// SGRLayer_64613488001024
// MI455X (gfx1250) — compile-verified, matrix operand order checked by reading, not run on hardware
//
#include <hip/hip_runtime.h>
#include <hip/hip_bf16.h>

#define D_DIM 1024
#define T_DIM 4096
#define B_DIM 4
#define ROWS (B_DIM * T_DIM)       // 16384 rows of length D
#define NCH  (B_DIM * D_DIM)       // 4096 independent scan channels
#define CHUNK 64                   // timesteps per scan chunk
#define NCHUNK (T_DIM / CHUNK)     // 64 chunks

typedef __attribute__((ext_vector_type(16))) __bf16 v16bf;
typedef __attribute__((ext_vector_type(8)))  float  v8f;

__device__ __forceinline__ unsigned short f32_to_bf16_rne(float f) {
    unsigned int u = __float_as_uint(f);
    u += 0x7FFFu + ((u >> 16) & 1u);   // round-to-nearest-even
    return (unsigned short)(u >> 16);
}

// ---------------------------------------------------------------------------
// Kernel 1: convert Wf and Wr (each D*D f32) to bf16
// ---------------------------------------------------------------------------
__global__ void convert_w_kernel(const float* __restrict__ Wf,
                                 const float* __restrict__ Wr,
                                 unsigned short* __restrict__ wf16,
                                 unsigned short* __restrict__ wr16) {
    int i = blockIdx.x * blockDim.x + threadIdx.x;
    const int N = D_DIM * D_DIM;
    if (i < N) {
        wf16[i] = f32_to_bf16_rne(Wf[i]);
    } else {
        int j = i - N;
        wr16[j] = f32_to_bf16_rne(Wr[j]);
    }
}

// ---------------------------------------------------------------------------
// Kernel 2: LayerNorm. One 256-thread block per row; 4 floats per thread.
// Writes fp32 x_norm (for the beta_x epilogue) and bf16 x_norm (WMMA A).
// ---------------------------------------------------------------------------
__global__ void ln_kernel(const float* __restrict__ x,
                          const float* __restrict__ gamma,
                          const float* __restrict__ beta,
                          float* __restrict__ xn_f32,
                          unsigned short* __restrict__ xn_bf16) {
    __shared__ float red_s[256];
    __shared__ float red_q[256];
    const int row = blockIdx.x;
    const int tid = threadIdx.x;

    const float4 v = ((const float4*)(x + (size_t)row * D_DIM))[tid];
    red_s[tid] = v.x + v.y + v.z + v.w;
    red_q[tid] = v.x * v.x + v.y * v.y + v.z * v.z + v.w * v.w;
    __syncthreads();
    for (int off = 128; off > 0; off >>= 1) {
        if (tid < off) {
            red_s[tid] += red_s[tid + off];
            red_q[tid] += red_q[tid + off];
        }
        __syncthreads();
    }
    const float inv_d = 1.0f / (float)D_DIM;
    const float mu  = red_s[0] * inv_d;
    const float var = red_q[0] * inv_d - mu * mu;
    const float rs  = rsqrtf(var + 1e-5f);

    const float4 g = ((const float4*)gamma)[tid];
    const float4 b = ((const float4*)beta)[tid];
    float4 o;
    o.x = (v.x - mu) * rs * g.x + b.x;
    o.y = (v.y - mu) * rs * g.y + b.y;
    o.z = (v.z - mu) * rs * g.z + b.z;
    o.w = (v.w - mu) * rs * g.w + b.w;
    ((float4*)(xn_f32 + (size_t)row * D_DIM))[tid] = o;

    uint2 pk;
    pk.x = (unsigned int)f32_to_bf16_rne(o.x) | ((unsigned int)f32_to_bf16_rne(o.y) << 16);
    pk.y = (unsigned int)f32_to_bf16_rne(o.z) | ((unsigned int)f32_to_bf16_rne(o.w) << 16);
    ((uint2*)(xn_bf16 + (size_t)row * D_DIM))[tid] = pk;
}

// ---------------------------------------------------------------------------
// WMMA 32x64 wave tile: 2 A-frags x 4 B-frags -> 8 accumulators.
// A row-major [M,K] bf16; Bm is W stored (out,in) row-major [N,K] bf16.
// Per CDNA5 16-bit A layout: lanes 0-15 K={0..7,16..23}, lanes 16-31
// K={8..15,24..31}; two b128 loads per lane per fragment.
// ---------------------------------------------------------------------------
__device__ __forceinline__ void wmma_gemm_32x64(const unsigned short* __restrict__ A,
                                                const unsigned short* __restrict__ Bm,
                                                int rowA0, int rowB0, int lane,
                                                v8f c[2][4]) {
    const int koff = (lane & 16) ? 8 : 0;
    const int r = lane & 15;
    const unsigned short* ap = A  + (size_t)(rowA0 + r) * D_DIM + koff;
    const unsigned short* bp = Bm + (size_t)(rowB0 + r) * D_DIM + koff;
    const size_t str16 = (size_t)16 * D_DIM;
#pragma unroll 2
    for (int k0 = 0; k0 < D_DIM; k0 += 32) {
        union { v16bf v; uint4 u[2]; } a[2], b[4];
#pragma unroll
        for (int mi = 0; mi < 2; ++mi) {
            a[mi].u[0] = *(const uint4*)(ap + mi * str16 + k0);
            a[mi].u[1] = *(const uint4*)(ap + mi * str16 + k0 + 16);
        }
#pragma unroll
        for (int ni = 0; ni < 4; ++ni) {
            b[ni].u[0] = *(const uint4*)(bp + ni * str16 + k0);
            b[ni].u[1] = *(const uint4*)(bp + ni * str16 + k0 + 16);
        }
#pragma unroll
        for (int mi = 0; mi < 2; ++mi)
#pragma unroll
            for (int ni = 0; ni < 4; ++ni)
                c[mi][ni] = __builtin_amdgcn_wmma_f32_16x16x32_bf16(
                        false, a[mi].v, false, b[ni].v,
                        (short)0, c[mi][ni], false, false);
    }
}

// ---------------------------------------------------------------------------
// Kernel 3: GEMM1 + gate epilogue.
// alpha = sigmoid(xn @ Wf^T + bf + alpha_bias)
// log_alpha = log(alpha + 1e-8)            -> la buffer
// beta_x    = beta_scale*(1-alpha)*xn_f32  -> overwrites xn_f32 in place
// Block = 8 waves (4M x 2N) -> 128x128 macro tile; wave tile 32x64.
// ---------------------------------------------------------------------------
__global__ void gemm1_kernel(const unsigned short* __restrict__ xn16,
                             const unsigned short* __restrict__ wf16,
                             const float* __restrict__ bf,
                             const float* __restrict__ alpha_bias,
                             const float* __restrict__ beta_scale,
                             float* __restrict__ xn_beta,   // in: xn f32, out: beta_x
                             float* __restrict__ log_alpha) {
    const int tid  = threadIdx.x;
    const int lane = tid & 31;
    const int wave = tid >> 5;
    const int rowA0 = blockIdx.x * 128 + (wave & 3) * 32;
    const int rowB0 = blockIdx.y * 128 + (wave >> 2) * 64;

    v8f c[2][4] = {};
    wmma_gemm_32x64(xn16, wf16, rowA0, rowB0, lane, c);

    const float ab = alpha_bias[0];
    const float bs = beta_scale[0];
    const int rsub = (lane & 16) ? 8 : 0;
    const int csub = lane & 15;
#pragma unroll
    for (int mi = 0; mi < 2; ++mi) {
#pragma unroll
        for (int ni = 0; ni < 4; ++ni) {
            const int col = rowB0 + ni * 16 + csub;
            const float bias = bf[col] + ab;
#pragma unroll
            for (int v = 0; v < 8; ++v) {
                const int row = rowA0 + mi * 16 + rsub + v;
                const size_t idx = (size_t)row * D_DIM + col;
                const float pre   = c[mi][ni][v] + bias;
                const float alpha = 1.0f / (1.0f + expf(-pre));
                log_alpha[idx] = logf(alpha + 1e-8f);
                xn_beta[idx]   = bs * (1.0f - alpha) * xn_beta[idx];
            }
        }
    }
}

// ---------------------------------------------------------------------------
// Chunked scans. Channel ch in [0,4096): b = ch>>10, d = ch&1023.
// Thread (chunk, ch): consecutive threads -> consecutive d -> coalesced.
// ---------------------------------------------------------------------------
__device__ __forceinline__ size_t chan_base(int ch, int chunk) {
    const int b = ch >> 10;
    const int d = ch & (D_DIM - 1);
    return (size_t)b * T_DIM * D_DIM + (size_t)chunk * CHUNK * D_DIM + d;
}

// K4a: per-chunk sums of log_alpha
__global__ void la_chunk_sum_kernel(const float* __restrict__ la, float* __restrict__ S) {
    const int idx = blockIdx.x * blockDim.x + threadIdx.x;   // 0..NCH*NCHUNK-1
    const int ch = idx & (NCH - 1);
    const int chunk = idx >> 12;
    const size_t base = chan_base(ch, chunk);
    float s = 0.0f;
    for (int t = 0; t < CHUNK; ++t) s += la[base + (size_t)t * D_DIM];
    S[idx] = s;
}

// K4b / K5b: per-channel exclusive prefix over NCHUNK chunk sums (in place)
__global__ void prefix_kernel(float* __restrict__ S) {
    const int ch = blockIdx.x * blockDim.x + threadIdx.x;    // 0..NCH-1
    float run = 0.0f;
    for (int cidx = 0; cidx < NCHUNK; ++cidx) {
        const int i = cidx * NCH + ch;
        const float v = S[i];
        S[i] = run;
        run += v;
    }
}

// K4c: in-place chunk cumsum with offset + per-chunk max
__global__ void la_cumsum_kernel(float* __restrict__ la, const float* __restrict__ S,
                                 float* __restrict__ cmax) {
    const int idx = blockIdx.x * blockDim.x + threadIdx.x;
    const int ch = idx & (NCH - 1);
    const int chunk = idx >> 12;
    const size_t base = chan_base(ch, chunk);
    float c = S[idx];
    float m = -3.402823466e38f;
    for (int t = 0; t < CHUNK; ++t) {
        const size_t i = base + (size_t)t * D_DIM;
        c += la[i];
        la[i] = c;
        m = fmaxf(m, c);
    }
    cmax[idx] = m;
}

// K4d: m[ch] = max over chunks
__global__ void max_reduce_kernel(const float* __restrict__ cmax, float* __restrict__ mbuf) {
    const int ch = blockIdx.x * blockDim.x + threadIdx.x;
    float m = -3.402823466e38f;
    for (int cidx = 0; cidx < NCHUNK; ++cidx) m = fmaxf(m, cmax[cidx * NCH + ch]);
    mbuf[ch] = m;
}

// K5a: per-chunk sums of x_star = beta_x * exp(cum - m)
__global__ void xs_chunk_sum_kernel(const float* __restrict__ cum,
                                    const float* __restrict__ mbuf,
                                    const float* __restrict__ bx,
                                    float* __restrict__ S2) {
    const int idx = blockIdx.x * blockDim.x + threadIdx.x;
    const int ch = idx & (NCH - 1);
    const int chunk = idx >> 12;
    const size_t base = chan_base(ch, chunk);
    const float m = mbuf[ch];
    float s = 0.0f;
    for (int t = 0; t < CHUNK; ++t) {
        const size_t i = base + (size_t)t * D_DIM;
        s += bx[i] * expf(cum[i] - m);
    }
    S2[idx] = s;
}

// K5c: s_star cumsum (with chunk offset) and output = s_star/(scale+eps) -> bf16
__global__ void scan_out_kernel(const float* __restrict__ cum,
                                const float* __restrict__ mbuf,
                                const float* __restrict__ bx,
                                const float* __restrict__ S2,
                                unsigned short* __restrict__ o16) {
    const int idx = blockIdx.x * blockDim.x + threadIdx.x;
    const int ch = idx & (NCH - 1);
    const int chunk = idx >> 12;
    const size_t base = chan_base(ch, chunk);
    const float m = mbuf[ch];
    float s = S2[idx];
    for (int t = 0; t < CHUNK; ++t) {
        const size_t i = base + (size_t)t * D_DIM;
        const float sc = expf(cum[i] - m);
        s += bx[i] * sc;
        o16[i] = f32_to_bf16_rne(s / (sc + 1e-8f));
    }
}

// ---------------------------------------------------------------------------
// Kernel 6: GEMM2 + tanh/residual epilogue.  out = 0.05*tanh(o @ Wr^T + br) + x
// ---------------------------------------------------------------------------
__global__ void gemm2_kernel(const unsigned short* __restrict__ o16,
                             const unsigned short* __restrict__ wr16,
                             const float* __restrict__ br,
                             const float* __restrict__ x,
                             float* __restrict__ out) {
    const int tid  = threadIdx.x;
    const int lane = tid & 31;
    const int wave = tid >> 5;
    const int rowA0 = blockIdx.x * 128 + (wave & 3) * 32;
    const int rowB0 = blockIdx.y * 128 + (wave >> 2) * 64;

    v8f c[2][4] = {};
    wmma_gemm_32x64(o16, wr16, rowA0, rowB0, lane, c);

    const int rsub = (lane & 16) ? 8 : 0;
    const int csub = lane & 15;
#pragma unroll
    for (int mi = 0; mi < 2; ++mi) {
#pragma unroll
        for (int ni = 0; ni < 4; ++ni) {
            const int col = rowB0 + ni * 16 + csub;
            const float bias = br[col];
#pragma unroll
            for (int v = 0; v < 8; ++v) {
                const int row = rowA0 + mi * 16 + rsub + v;
                const size_t idx = (size_t)row * D_DIM + col;
                out[idx] = 0.05f * tanhf(c[mi][ni][v] + bias) + x[idx];
            }
        }
    }
}

// ---------------------------------------------------------------------------
extern "C" void kernel_launch(void* const* d_in, const int* in_sizes, int n_in,
                              void* d_out, int out_size, void* d_ws, size_t ws_size,
                              hipStream_t stream) {
    const float* x          = (const float*)d_in[0];
    const float* ln_w       = (const float*)d_in[1];
    const float* ln_b       = (const float*)d_in[2];
    const float* Wf         = (const float*)d_in[3];
    const float* bf         = (const float*)d_in[4];
    const float* Wr         = (const float*)d_in[5];
    const float* br         = (const float*)d_in[6];
    const float* alpha_bias = (const float*)d_in[7];
    const float* beta_scale = (const float*)d_in[8];
    float* out = (float*)d_out;

    char* ws = (char*)d_ws;
    size_t off = 0;
    float* xn_f32 = (float*)(ws + off);                 off += (size_t)ROWS * D_DIM * 4;   // 64 MB; becomes beta_x
    unsigned short* xn16 = (unsigned short*)(ws + off); off += (size_t)ROWS * D_DIM * 2;   // 32 MB
    float* la = (float*)(ws + off);                     off += (size_t)ROWS * D_DIM * 4;   // 64 MB; becomes cumsum
    unsigned short* o16 = (unsigned short*)(ws + off);  off += (size_t)ROWS * D_DIM * 2;   // 32 MB
    float* mbuf = (float*)(ws + off);                   off += (size_t)NCH * 4;            // 16 KB
    unsigned short* wf16 = (unsigned short*)(ws + off); off += (size_t)D_DIM * D_DIM * 2;  // 2 MB
    unsigned short* wr16 = (unsigned short*)(ws + off); off += (size_t)D_DIM * D_DIM * 2;  // 2 MB
    float* S    = (float*)(ws + off);                   off += (size_t)NCH * NCHUNK * 4;   // 1 MB
    float* S2   = (float*)(ws + off);                   off += (size_t)NCH * NCHUNK * 4;   // 1 MB
    float* cmax = (float*)(ws + off);                   off += (size_t)NCH * NCHUNK * 4;   // 1 MB

    convert_w_kernel<<<(2 * D_DIM * D_DIM) / 256, 256, 0, stream>>>(Wf, Wr, wf16, wr16);
    ln_kernel<<<ROWS, 256, 0, stream>>>(x, ln_w, ln_b, xn_f32, xn16);

    dim3 ggrid(ROWS / 128, D_DIM / 128);
    gemm1_kernel<<<ggrid, 256, 0, stream>>>(xn16, wf16, bf, alpha_bias, beta_scale, xn_f32, la);

    const int scan_blocks = (NCH * NCHUNK) / 256;   // 1024
    const int ch_blocks   = NCH / 256;              // 16
    la_chunk_sum_kernel<<<scan_blocks, 256, 0, stream>>>(la, S);
    prefix_kernel<<<ch_blocks, 256, 0, stream>>>(S);
    la_cumsum_kernel<<<scan_blocks, 256, 0, stream>>>(la, S, cmax);
    max_reduce_kernel<<<ch_blocks, 256, 0, stream>>>(cmax, mbuf);
    xs_chunk_sum_kernel<<<scan_blocks, 256, 0, stream>>>(la, mbuf, xn_f32, S2);
    prefix_kernel<<<ch_blocks, 256, 0, stream>>>(S2);
    scan_out_kernel<<<scan_blocks, 256, 0, stream>>>(la, mbuf, xn_f32, S2, o16);

    gemm2_kernel<<<ggrid, 256, 0, stream>>>(o16, wr16, br, x, out);
}
